// MultiheadAttention_10763188044553
// MI455X (gfx1250) — matrix operand order repaired and checked by reading, not compiled, not run on hardware
//
#include <hip/hip_runtime.h>

// ---------------------------------------------------------------------------
// Multi-head attention forward for MI455X (gfx1250), bf16 WMMA + f32 accum.
//   out   = (softmax(QK^T/sqrt(dh) + bias) V) W_o        [B,L,D]  f32
//   align = softmax-weights transposed                   [B,H,Lk,Lq] f32
// Roofline: 69 GFLOP vs ~1GB mandatory traffic -> bf16 WMMA is HBM-bound.
// GEMMs stage the shared B-panel in LDS via async-to-LDS (ASYNCcnt) double
// buffering; attention uses 128KB of the 320KB WGP LDS for full-row softmax.
// ---------------------------------------------------------------------------

#define B_ 2
#define L_ 2048
#define D_ 1024
#define H_ 16
#define DH_ 64
#define N3_ 3072   // 3*D

typedef __bf16 bf16_t;
typedef __attribute__((ext_vector_type(16))) __bf16 v16bf;
typedef __attribute__((ext_vector_type(8)))  __bf16 bf16x8;
typedef __attribute__((ext_vector_type(8)))  float  v8f;
typedef __attribute__((ext_vector_type(4)))  float  f32x4;
typedef __attribute__((ext_vector_type(4)))  int    v4i;

#if __has_builtin(__builtin_amdgcn_global_load_async_to_lds_b128)
#define HAVE_ASYNC_LDS 1
#endif

// Builtin expects int4* in explicit address spaces (as1 = global, as3 = LDS).
typedef __attribute__((address_space(1))) v4i* gv4i_p;
typedef __attribute__((address_space(3))) v4i* lv4i_p;

// Copy 16B global -> LDS. Async (ASYNCcnt-tracked) when available.
__device__ __forceinline__ void async_copy16(const bf16_t* g, bf16_t* l) {
#ifdef HAVE_ASYNC_LDS
  __builtin_amdgcn_global_load_async_to_lds_b128((gv4i_p)g, (lv4i_p)l, 0, 0);
#else
  *(bf16x8*)l = *(const bf16x8*)g;
#endif
}

template <int N>
__device__ __forceinline__ void wait_async() {
#ifdef HAVE_ASYNC_LDS
#if __has_builtin(__builtin_amdgcn_s_wait_asynccnt)
  __builtin_amdgcn_s_wait_asynccnt(N);
#else
  asm volatile("s_wait_asynccnt %0" ::"i"(N) : "memory");
#endif
#endif
}

__device__ __forceinline__ v8f wmma_bf16(v16bf a, v16bf b, v8f c) {
  return __builtin_amdgcn_wmma_f32_16x16x32_bf16(false, a, false, b,
                                                 (short)0, c, false, false);
}

// A-operand (16x32 bf16, MxK): lane r=l&15 holds row M=r, half=l>>4 selects the
// K-interleave: two contiguous 8-elem runs at k = 8*half and k = 16 + 8*half.
__device__ __forceinline__ v16bf load_a16(const bf16_t* rowp, int kb, int hf) {
  bf16x8 lo = *(const bf16x8*)(rowp + kb + hf * 8);
  bf16x8 hi = *(const bf16x8*)(rowp + kb + 16 + hf * 8);
  v16bf r;
#pragma unroll
  for (int i = 0; i < 8; ++i) { r[i] = lo[i]; r[i + 8] = hi[i]; }
  return r;
}

// B-operand (32x16 bf16, KxN): lane holds column N=l&15; half=l>>4 selects the
// K-interleave: two contiguous 8-elem runs at k = 8*half and k = 16 + 8*half.
// rowp points at B^T row n (contiguous in K).
__device__ __forceinline__ v16bf load_b16(const bf16_t* rowp, int kb, int hf) {
  const bf16_t* p = rowp + kb + hf * 8;
  bf16x8 lo = *(const bf16x8*)(p);
  bf16x8 hi = *(const bf16x8*)(p + 16);
  v16bf r;
#pragma unroll
  for (int i = 0; i < 8; ++i) { r[i] = lo[i]; r[i + 8] = hi[i]; }
  return r;
}

// Stage one 64x32 bf16 B-panel (rows n0..n0+63 of B^T, k range [kb,kb+32)) to
// LDS: 256 threads x 16B = 4KB. LDS layout: [n_local][32] bf16 row-major.
__device__ __forceinline__ void stage_B(const bf16_t* Bt, int K, int n0, int kb,
                                        bf16_t* dst, int tid) {
  int nl = tid >> 2, q4 = tid & 3;
  async_copy16(Bt + (size_t)(n0 + nl) * K + kb + q4 * 8, dst + nl * 32 + q4 * 8);
}

// Block GEMM core: 8 waves, wave w owns rows m0..m0+15, all waves share the
// LDS-staged B-panel (64 cols), double-buffered async pipeline over K.
__device__ __forceinline__ void gemm_core_ldsB(const bf16_t* A,
                                               const bf16_t* Bt, int K, int m0,
                                               int n0, bf16_t* ldsB, int tid,
                                               int lane, v8f acc[4]) {
  const int hf = lane >> 4, r = lane & 15;
  const bf16_t* pa = A + (size_t)(m0 + r) * K;
  const int nIter = K / 32;
  stage_B(Bt, K, n0, 0, ldsB, tid);
  for (int i = 0; i < nIter; ++i) {
    bf16_t* cur = ldsB + (i & 1) * (64 * 32);
    if (i + 1 < nIter) {
      stage_B(Bt, K, n0, (i + 1) * 32, ldsB + ((i + 1) & 1) * (64 * 32), tid);
      wait_async<1>();   // stage i landed, stage i+1 still in flight
    } else {
      wait_async<0>();
    }
    __syncthreads();
    v16bf a = load_a16(pa, i * 32, hf);
#pragma unroll
    for (int t = 0; t < 4; ++t)
      acc[t] = wmma_bf16(a, load_b16(cur + (t * 16 + r) * 32, 0, hf), acc[t]);
    __syncthreads();   // everyone done with `cur` before it is overwritten
  }
}

// ---------------------------------------------------------------------------
// Kernel 0a: f32 -> bf16 copy (queries)
__global__ void mha_conv_cast(const float* __restrict__ src,
                              bf16_t* __restrict__ dst, int total) {
  for (int i = blockIdx.x * blockDim.x + threadIdx.x; i < total;
       i += gridDim.x * blockDim.x)
    dst[i] = (bf16_t)src[i];
}

// Kernel 0b: f32 [K][N] -> bf16 transposed [N][K]
__global__ void mha_conv_transpose(const float* __restrict__ src,
                                   bf16_t* __restrict__ dst, int K, int N) {
  int total = K * N;
  for (int i = blockIdx.x * blockDim.x + threadIdx.x; i < total;
       i += gridDim.x * blockDim.x) {
    int n = i / K, k = i - n * K;
    dst[i] = (bf16_t)src[(size_t)k * N + n];
  }
}

// ---------------------------------------------------------------------------
// Kernel 1: QKV projection. C[4096x3072] = Qbf x Wqkv, scattered into
// q_h[b][h][l][dh] (scaled by DH^-0.5), k_h[b][h][l][dh], v_t[b][h][dh][l].
// Grid: (4096/128) m-blocks x 48 n-groups.
__global__ __launch_bounds__(256) void mha_gemm_qkv(
    const bf16_t* __restrict__ Qbf, const bf16_t* __restrict__ WqkvT,
    bf16_t* __restrict__ q_h, bf16_t* __restrict__ k_h,
    bf16_t* __restrict__ v_t) {
  __shared__ bf16_t ldsB[2][64 * 32];
  const int NG = N3_ / 64;  // 48
  int mb = blockIdx.x / NG, ng = blockIdx.x - mb * NG;
  int tid = threadIdx.x, wave = tid >> 5, lane = tid & 31;
  int m0 = mb * 128 + wave * 16, n0 = ng * 64;

  v8f acc[4] = {};
  gemm_core_ldsB(Qbf, WqkvT, D_, m0, n0, &ldsB[0][0], tid, lane, acc);

  int sec = n0 / D_;            // 0=q, 1=k, 2=v (64-col group never straddles)
  int h = (n0 % D_) / DH_;
  int hf = lane >> 4, nn = lane & 15;
#pragma unroll
  for (int t = 0; t < 4; ++t) {
    int dh = t * 16 + nn;
#pragma unroll
    for (int r8 = 0; r8 < 8; ++r8) {
      int m = m0 + r8 + 8 * hf;
      int b = m >> 11, lq = m & (L_ - 1);
      size_t bh = (size_t)(b * H_ + h);
      float v = acc[t][r8];
      if (sec == 0)
        q_h[(bh * L_ + lq) * DH_ + dh] = (bf16_t)(v * 0.125f);  // DH^-0.5
      else if (sec == 1)
        k_h[(bh * L_ + lq) * DH_ + dh] = (bf16_t)v;
      else
        v_t[(bh * DH_ + dh) * L_ + lq] = (bf16_t)v;
    }
  }
}

// ---------------------------------------------------------------------------
// Kernel 2: attention. One block per (b, h, 16 query rows).
// LDS: logits/weights [16][2048] f32 (128KB), red[256], ctxs[1024].
__global__ __launch_bounds__(256) void mha_attention(
    const bf16_t* __restrict__ q_h, const bf16_t* __restrict__ k_h,
    const bf16_t* __restrict__ v_t, const float* __restrict__ bias,
    float* __restrict__ align, bf16_t* __restrict__ x_bf) {
  extern __shared__ float sm[];          // [16 * 2048]
  float* red  = sm + 16 * L_;            // [16 * 16]
  float* ctxs = red + 256;               // [4 * 256]

  const int tid = threadIdx.x;
  const int wave = tid >> 5, lane = tid & 31;
  const int hf = lane >> 4, r = lane & 15;
  const int bid = blockIdx.x;
  const int qt = bid & 127, h = (bid >> 7) & 15, b = bid >> 11;
  const int q0 = qt * 16;
  const size_t bh = (size_t)(b * H_ + h);

  // ---- Phase 1: logits = q (16 x 64) @ k^T (64 x 2048), + bias, into LDS.
  const bf16_t* pq = q_h + (bh * L_ + q0 + r) * DH_;
  v16bf aq0 = load_a16(pq, 0, hf);
  v16bf aq1 = load_a16(pq, 32, hf);
  for (int kt = wave; kt < L_ / 16; kt += 8) {
    int kk = kt * 16 + r;                         // B column = key index
    const bf16_t* pk = k_h + (bh * L_ + kk) * DH_;
    if (kt + 8 < L_ / 16)
      __builtin_prefetch(k_h + (bh * L_ + kk + 128) * DH_, 0, 0);
    v8f acc = {};
    acc = wmma_bf16(aq0, load_b16(pk, 0, hf), acc);
    acc = wmma_bf16(aq1, load_b16(pk, 32, hf), acc);
    float bv = bias[b * L_ + kk];
#pragma unroll
    for (int r8 = 0; r8 < 8; ++r8)
      sm[(r8 + 8 * hf) * L_ + kk] = acc[r8] + bv;
  }
  __syncthreads();

  // ---- Phase 2: row softmax over 2048 cols (16 threads per row).
  const int row = tid >> 4, sub = tid & 15;
  float* srow = sm + row * L_;
  float mx = -3.402823466e38f;
  for (int c = sub; c < L_; c += 16) mx = fmaxf(mx, srow[c]);
  red[row * 16 + sub] = mx;
  __syncthreads();
  if (sub == 0) {
    float m2 = red[row * 16];
#pragma unroll
    for (int i = 1; i < 16; ++i) m2 = fmaxf(m2, red[row * 16 + i]);
    red[row * 16] = m2;
  }
  __syncthreads();
  mx = red[row * 16];
  __syncthreads();
  float s = 0.f;
  for (int c = sub; c < L_; c += 16) {
    float e = __expf(srow[c] - mx);
    srow[c] = e;
    s += e;
  }
  red[row * 16 + sub] = s;
  __syncthreads();
  if (sub == 0) {
    float t2 = 0.f;
#pragma unroll
    for (int i = 0; i < 16; ++i) t2 += red[row * 16 + i];
    red[row * 16] = 1.0f / t2;
  }
  __syncthreads();
  float rinv = red[row * 16];
  for (int c = sub; c < L_; c += 16) srow[c] *= rinv;
  __syncthreads();

  // ---- Phase 3: align[b][h][kk][q] = weights[q][kk] (transposed stream-out).
  {
    int qoff = tid & 15, k0 = tid >> 4;
    for (int kk = k0; kk < L_; kk += 16)
      align[(bh * L_ + kk) * L_ + q0 + qoff] = sm[qoff * L_ + kk];
  }

  // ---- Phase 4: context = weights (16x2048) @ v (2048x64).
  // 8 waves: dh-tile = wave&3, K-half = wave>>2; reduce halves through LDS.
  const int dhT = wave & 3, khalf = wave >> 2;
  const int dh = dhT * 16 + r;                    // B column = dh
  const bf16_t* pv = v_t + (bh * DH_ + dh) * L_;
  const float* wrow = sm + r * L_;                // A row = local q
  v8f acc = {};
  for (int kb = khalf * 1024; kb < khalf * 1024 + 1024; kb += 32) {
    const float* p0 = wrow + kb + hf * 8;
    const float* p1 = wrow + kb + 16 + hf * 8;
    f32x4 a0 = *(const f32x4*)(p0);
    f32x4 a1 = *(const f32x4*)(p0 + 4);
    f32x4 a2 = *(const f32x4*)(p1);
    f32x4 a3 = *(const f32x4*)(p1 + 4);
    v16bf aw;
#pragma unroll
    for (int i = 0; i < 4; ++i) {
      aw[i]      = (bf16_t)a0[i];
      aw[4 + i]  = (bf16_t)a1[i];
      aw[8 + i]  = (bf16_t)a2[i];
      aw[12 + i] = (bf16_t)a3[i];
    }
    acc = wmma_bf16(aw, load_b16(pv, kb, hf), acc);
  }
  if (khalf == 1) {
#pragma unroll
    for (int r8 = 0; r8 < 8; ++r8) ctxs[dhT * 256 + r8 * 32 + lane] = acc[r8];
  }
  __syncthreads();
  if (khalf == 0) {
#pragma unroll
    for (int r8 = 0; r8 < 8; ++r8) {
      float v = acc[r8] + ctxs[dhT * 256 + r8 * 32 + lane];
      int q = q0 + r8 + 8 * hf;
      x_bf[(size_t)(b * L_ + q) * D_ + h * DH_ + dh] = (bf16_t)v;
    }
  }
}

// ---------------------------------------------------------------------------
// Kernel 3: out[4096x1024] = X_bf x W_o, f32 store to d_out.
// Grid: (4096/128) m-blocks x 16 n-groups.
__global__ __launch_bounds__(256) void mha_gemm_out(
    const bf16_t* __restrict__ Xbf, const bf16_t* __restrict__ WoT,
    float* __restrict__ out) {
  __shared__ bf16_t ldsB[2][64 * 32];
  const int NG = D_ / 64;  // 16
  int mb = blockIdx.x / NG, ng = blockIdx.x - mb * NG;
  int tid = threadIdx.x, wave = tid >> 5, lane = tid & 31;
  int m0 = mb * 128 + wave * 16, n0 = ng * 64;

  v8f acc[4] = {};
  gemm_core_ldsB(Xbf, WoT, D_, m0, n0, &ldsB[0][0], tid, lane, acc);

  int hf = lane >> 4, nn = lane & 15;
#pragma unroll
  for (int t = 0; t < 4; ++t)
#pragma unroll
    for (int r8 = 0; r8 < 8; ++r8)
      out[(size_t)(m0 + r8 + 8 * hf) * D_ + n0 + t * 16 + nn] = acc[t][r8];
}

// ---------------------------------------------------------------------------
extern "C" void kernel_launch(void* const* d_in, const int* in_sizes, int n_in,
                              void* d_out, int out_size, void* d_ws,
                              size_t ws_size, hipStream_t stream) {
  const float* queries = (const float*)d_in[0];  // [B,L,D]
  const float* bias    = (const float*)d_in[1];  // [B,1,1,L]
  const float* w_qkv   = (const float*)d_in[2];  // [D,3D]
  const float* w_o     = (const float*)d_in[3];  // [D,D]

  float* out   = (float*)d_out;                        // [B,L,D]
  float* align = out + (size_t)B_ * L_ * D_;           // [B,H,L,L]

  // Workspace layout (bf16 buffers, ~48 MB total).
  char* ws = (char*)d_ws;
  size_t o = 0;
  bf16_t* Qbf   = (bf16_t*)(ws + o); o += (size_t)B_ * L_ * D_ * 2;     // 8MB
  bf16_t* WqkvT = (bf16_t*)(ws + o); o += (size_t)N3_ * D_ * 2;         // 6MB
  bf16_t* WoT   = (bf16_t*)(ws + o); o += (size_t)D_ * D_ * 2;          // 2MB
  bf16_t* q_h   = (bf16_t*)(ws + o); o += (size_t)B_ * H_ * L_ * DH_ * 2;
  bf16_t* k_h   = (bf16_t*)(ws + o); o += (size_t)B_ * H_ * L_ * DH_ * 2;
  bf16_t* v_t   = (bf16_t*)(ws + o); o += (size_t)B_ * H_ * L_ * DH_ * 2;
  bf16_t* x_bf  = (bf16_t*)(ws + o); o += (size_t)B_ * L_ * D_ * 2;

  // 0) precision conversion / weight transposes
  mha_conv_cast<<<1024, 256, 0, stream>>>(queries, Qbf, B_ * L_ * D_);
  mha_conv_transpose<<<1024, 256, 0, stream>>>(w_qkv, WqkvT, D_, N3_);
  mha_conv_transpose<<<512, 256, 0, stream>>>(w_o, WoT, D_, D_);

  // 1) QKV projection GEMM: 32 m-blocks x 48 n-groups
  mha_gemm_qkv<<<32 * 48, 256, 0, stream>>>(Qbf, WqkvT, q_h, k_h, v_t);

  // 2) attention (4096 blocks, 136KB dynamic LDS each)
  size_t lds_bytes = (size_t)(16 * L_ + 256 + 1024) * sizeof(float);
  mha_attention<<<B_ * H_ * (L_ / 16), 256, lds_bytes, stream>>>(
      q_h, k_h, v_t, bias, align, x_bf);

  // 3) output projection GEMM: 32 m-blocks x 16 n-groups
  mha_gemm_out<<<32 * 16, 256, 0, stream>>>(x_bf, WoT, out);
}
